// DIFMultiHeadAttention_16415365005499
// MI455X (gfx1250) — matrix operand order repaired and checked by reading, not compiled, not run on hardware
//
#include <hip/hip_runtime.h>
#include <hip/hip_bf16.h>

// ---------------------------------------------------------------------------
// CDNA5 (gfx1250) fused DIF multi-head attention.
// bf16 WMMA pipeline, bf16 intermediate activations (halves HBM traffic),
// TDM (tensor_load_to_lds) weight staging, vectorized b128 fragment loads.
//   B=64, L=200, D=256, H=8, F=2, d=32
// ---------------------------------------------------------------------------

#define Bb 64
#define Ll 200
#define Dd 256
#define Hh 8
#define Ff 2
#define DH 32
#define LP 224            // L padded to multiple of 32 (224 = 7*32)
#define MROWS (Bb * Ll)   // 12800

typedef __attribute__((ext_vector_type(16))) __bf16 v16bf;
typedef __attribute__((ext_vector_type(8)))  __bf16 v8bf;
typedef __attribute__((ext_vector_type(8)))  float  v8f;
typedef __attribute__((ext_vector_type(4)))  unsigned int u32x4;
typedef __attribute__((ext_vector_type(8)))  int i32x8;
typedef __attribute__((ext_vector_type(4)))  int i32x4;

// --------------------------- WMMA wrapper ----------------------------------
__device__ __forceinline__ v8f wmma_bf(v16bf a, v16bf b, v8f c) {
  return __builtin_amdgcn_wmma_f32_16x16x32_bf16(
      false, a, false, b, (short)0, c, false, false);
}

// ------------------- TDM: 2D f32 tile -> LDS (ISA ch.8 D#) -----------------
// D# group0: count=1 | lds_addr | global_addr[56:0] | type=2 ("image")
// D# group1: data_size=4B | tensor_dim0/1 | tile_dim0/1 | tensor_dim0_stride
__device__ __forceinline__ void tdm_load_tile_f32(
    const float* gsrc, unsigned lds_off,
    unsigned tile_w, unsigned tile_h,
    unsigned tensor_w, unsigned tensor_h, unsigned row_stride) {
  unsigned long long ga = (unsigned long long)(uintptr_t)gsrc;
  u32x4 g0 = {1u,                                  // count=1, user mode
              lds_off,                             // LDS byte address
              (unsigned)(ga & 0xffffffffu),
              (unsigned)((ga >> 32) & 0x01ffffffu) | 0x80000000u}; // type=2
  i32x8 g1 = {(int)0x00020000u,                    // data_size = 4B
              (int)((tensor_w & 0xffffu) << 16),   // tensor_dim0[15:0]
              (int)(((tensor_w >> 16) & 0xffffu) | ((tensor_h & 0xffffu) << 16)),
              (int)(((tensor_h >> 16) & 0xffffu) | ((tile_w & 0xffffu) << 16)),
              (int)(tile_h & 0xffffu),             // tile_dim1 (tile_dim2=0)
              (int)row_stride,                     // tensor_dim0_stride[31:0]
              0, 0};
  i32x4 z4 = {0, 0, 0, 0};
#if defined(__clang_major__) && __clang_major__ >= 23
  i32x8 z8 = {0, 0, 0, 0, 0, 0, 0, 0};
  __builtin_amdgcn_tensor_load_to_lds(g0, g1, z4, z4, z8, 0);
#else
  __builtin_amdgcn_tensor_load_to_lds(g0, g1, z4, z4, 0);
#endif
}

// ---------------- Fragment loaders (contiguous k order, same in A and B) --------------------
// A (16x32 bf16): lanes 0-15 row=lane, elems 0..15 -> K 0..15;
// lanes 16-31 row=lane-16, elems -> K 16..31. Two 16B vector loads.
__device__ __forceinline__ v16bf a_frag_bf16(const __bf16* __restrict__ A,
                                             int lda, long r0, int k0) {
  int lane = threadIdx.x & 31;
  long m   = r0 + (lane & 15);
  const __bf16* p = A + m * lda + k0 + (lane >> 4) * 16;
  v8bf lo = *(const v8bf*)(p);        // K +0..7 (this half-wave)
  v8bf hi = *(const v8bf*)(p + 8);    // K +8..15
  return __builtin_shufflevector(lo, hi, 0, 1, 2, 3, 4, 5, 6, 7,
                                 8, 9, 10, 11, 12, 13, 14, 15);
}

__device__ __forceinline__ v16bf a_frag_lds_bf16(const __bf16* A, int lda,
                                                 int r0, int k0) {
  int lane = threadIdx.x & 31;
  int m    = r0 + (lane & 15);
  const __bf16* p = A + m * lda + k0 + (lane >> 4) * 16;
  v8bf lo = *(const v8bf*)(p);
  v8bf hi = *(const v8bf*)(p + 8);
  return __builtin_shufflevector(lo, hi, 0, 1, 2, 3, 4, 5, 6, 7,
                                 8, 9, 10, 11, 12, 13, 14, 15);
}

// A (16x32) from f32 source, converted to bf16 (projection inputs).
__device__ __forceinline__ v16bf a_frag_f32(const float* __restrict__ A,
                                            int lda, long r0, int k0) {
  int lane = threadIdx.x & 31;
  long m   = r0 + (lane & 15);
  const float4* p = (const float4*)(A + m * lda + k0 + (lane >> 4) * 16);
  float4 x0 = p[0], x1 = p[1];        // K +0..7
  float4 y0 = p[2], y1 = p[3];        // K +8..15
  v16bf o;
  o[0] = (__bf16)x0.x; o[1] = (__bf16)x0.y; o[2] = (__bf16)x0.z; o[3] = (__bf16)x0.w;
  o[4] = (__bf16)x1.x; o[5] = (__bf16)x1.y; o[6] = (__bf16)x1.z; o[7] = (__bf16)x1.w;
  o[8]  = (__bf16)y0.x; o[9]  = (__bf16)y0.y; o[10] = (__bf16)y0.z; o[11] = (__bf16)y0.w;
  o[12] = (__bf16)y1.x; o[13] = (__bf16)y1.y; o[14] = (__bf16)y1.z; o[15] = (__bf16)y1.w;
  return o;
}

// B (32x16): lane holds column n0+(lane&15); lanes 0-15 K=k0..15, 16-31 K=+16..31.
// Source pre-transposed: S[n][k] row-major -> 16 contiguous bf16 per lane.
__device__ __forceinline__ v16bf bT_frag_bf16(const __bf16* __restrict__ S,
                                              int lds_, int n0, int k0) {
  int lane = threadIdx.x & 31;
  int n    = n0 + (lane & 15);
  const __bf16* p = S + (long)n * lds_ + k0 + (lane >> 4) * 16;
  v8bf lo = *(const v8bf*)(p);
  v8bf hi = *(const v8bf*)(p + 8);
  return __builtin_shufflevector(lo, hi, 0, 1, 2, 3, 4, 5, 6, 7,
                                 8, 9, 10, 11, 12, 13, 14, 15);
}

// B (32x16) from row-major [K x N] bf16 source (strided columns).
__device__ __forceinline__ v16bf b_frag_bf16(const __bf16* __restrict__ Bm,
                                             int ldb, int k0, int n0) {
  int lane = threadIdx.x & 31;
  int n    = n0 + (lane & 15);
  int kb   = k0 + (lane >> 4) * 16;
  v16bf o;
#pragma unroll
  for (int e = 0; e < 16; ++e) o[e] = Bm[(long)(kb + e) * ldb + n];
  return o;
}

// B (32x16) from f32 tile in LDS (TDM-staged weights), converted to bf16.
__device__ __forceinline__ v16bf b_frag_lds_f32(const float* Bm, int ldb,
                                                int k0, int n0) {
  int lane = threadIdx.x & 31;
  int n    = n0 + (lane & 15);
  int kb   = k0 + (lane >> 4) * 16;
  v16bf o;
#pragma unroll
  for (int e = 0; e < 16; ++e) o[e] = (__bf16)Bm[(kb + e) * ldb + n];
  return o;
}

// ===========================================================================
// Prep kernels: pad/transpose weights to bf16 so hot loops are checkless.
// ===========================================================================
__global__ __launch_bounds__(256) void prep_gate_kernel(
    const float* __restrict__ Wg1, const float* __restrict__ bg1,
    const float* __restrict__ Wg2, __bf16* __restrict__ Wg1t,
    float* __restrict__ bg1p, float* __restrict__ wg2p) {
  int idx = blockIdx.x * 256 + threadIdx.x;
  if (idx < LP * LP) {
    int n = idx / LP, k = idx % LP;          // Wg1t[n][k] = Wg1[k][n]
    Wg1t[idx] = (__bf16)((n < Ll && k < Ll) ? Wg1[k * Ll + n] : 0.0f);
  }
  if (idx < LP) {
    bg1p[idx] = (idx < Ll) ? bg1[idx] : 0.0f;
    wg2p[idx] = (idx < Ll) ? Wg2[idx] : 0.0f;
  }
}

__global__ __launch_bounds__(256) void prep_wd_kernel(
    const float* __restrict__ Wd, __bf16* __restrict__ WdT) {
  int idx = blockIdx.x * 256 + threadIdx.x;  // 256*256
  int n = idx >> 8, k = idx & 255;           // WdT[n][k] = Wd[k][n]
  WdT[idx] = (__bf16)Wd[k * Dd + n];
}

// ===========================================================================
// Kernel 1: projection GEMM  C[M,256](bf16) = A[M,256](f32) @ W + bias
// 8 waves x 16 rows = 128 rows x 16 cols per block. Weight tile (32x16 f32)
// staged by the Tensor Data Mover into LDS each k-step, s_wait_tensorcnt.
// ===========================================================================
__global__ __launch_bounds__(256) void gemm_bias_kernel(
    const float* __restrict__ A, const float* __restrict__ W,
    const float* __restrict__ bias, __bf16* __restrict__ C) {
  __shared__ float wtile[32 * 16];
  int wave = threadIdx.x >> 5;
  int lane = threadIdx.x & 31;
  int n0   = blockIdx.x * 16;
  long r0  = (long)blockIdx.y * 128 + wave * 16;

  v8f acc = {};
#pragma unroll
  for (int k0 = 0; k0 < Dd; k0 += 32) {
    __syncthreads();                          // previous tile consumed
    if (wave == 0) {
      tdm_load_tile_f32(W + (long)k0 * Dd + n0,
                        (unsigned)(uintptr_t)&wtile[0],   // LDS aperture low bits
                        /*tile_w=*/16, /*tile_h=*/32,
                        /*tensor_w=*/Dd, /*tensor_h=*/Dd, /*stride=*/Dd);
      __builtin_amdgcn_s_wait_tensorcnt(0);
    }
    __syncthreads();                          // tile visible to all waves
    if (k0 + 32 < Dd)
      __builtin_prefetch(&A[r0 * Dd + k0 + 32], 0, 1);
    acc = wmma_bf(a_frag_f32(A, Dd, r0, k0), b_frag_lds_f32(wtile, 16, 0, 0), acc);
  }

  int col  = n0 + (lane & 15);
  int hw   = lane >> 4;
  float bv = bias[col];
#pragma unroll
  for (int r = 0; r < 8; ++r)
    C[(r0 + r + 8 * hw) * Dd + col] = (__bf16)(acc[r] + bv);
}

// ===========================================================================
// Kernel 2: fused attention core per (b, h, 16-row q tile). All four score
// streams live in LDS; gate GEMM, both softmaxes and probs@v never touch HBM
// for any [L,L] tensor.
// ===========================================================================
__global__ __launch_bounds__(256) void attn_fused_kernel(
    const __bf16* __restrict__ Q,  const __bf16* __restrict__ Km,
    const __bf16* __restrict__ V,  const __bf16* __restrict__ QP,
    const __bf16* __restrict__ KP, const __bf16* __restrict__ QA,
    const __bf16* __restrict__ KA, const float* __restrict__ mask,
    const __bf16* __restrict__ Wg1t, const float* __restrict__ bg1p,
    const float* __restrict__ wg2p,  const float* __restrict__ bg2v,
    __bf16* __restrict__ CTXb) {
  __shared__ __bf16 comb[4 * 16 * LP];
  __shared__ __bf16 probsb[16 * LP];
  __shared__ float  scores[16 * LP];
  __shared__ float  energy[64];
  __shared__ float  gw[64];
  __shared__ float  red[256];
  __shared__ float  rowstat[16];

  const int tid  = threadIdx.x;
  const int wave = tid >> 5;
  const int lane = tid & 31;
  const int hw   = lane >> 4;
  const int b    = blockIdx.z;
  const int h    = blockIdx.y;
  const int q0   = blockIdx.x * 16;

  const long rowBase = (long)b * Ll;
  const long FSTR    = (long)MROWS * Dd;
  const int  hcol    = h * DH;

  for (int i = tid; i < 64; i += 256) energy[i] = 0.0f;

  // ---- Stage 1: four score streams (one WMMA each tile, K = d = 32).
  // jobs = 4 streams x 14 key tiles = 56 -> 7 per wave (uniform).
#pragma unroll
  for (int t = 0; t < 7; ++t) {
    int job = wave + t * 8;
    int s = job / (LP / 16);
    int j = job % (LP / 16);
    const __bf16* qsrc;
    const __bf16* ksrc;
    if      (s == 0) { qsrc = QA;        ksrc = KA;        }
    else if (s == 1) { qsrc = QA + FSTR; ksrc = KA + FSTR; }
    else if (s == 2) { qsrc = Q;         ksrc = Km;        }
    else             { qsrc = QP;        ksrc = KP;        }
    v8f z = {};
    v8f acc = wmma_bf(a_frag_bf16(qsrc, Dd, rowBase + q0, hcol),
                      bT_frag_bf16(ksrc + rowBase * Dd + hcol, Dd, j * 16, 0), z);
    int col = j * 16 + (lane & 15);
    bool colValid = (j * 16 + (lane & 15)) < Ll;
#pragma unroll
    for (int r = 0; r < 8; ++r)
      comb[(s * 16 + r + 8 * hw) * LP + col] = colValid ? (__bf16)acc[r] : (__bf16)0.0f;
  }
  __syncthreads();

  // ---- Stage 2: gate GEMM  relu(comb @ Wg1 + bg1) . Wg2 -> energy[64]
#pragma unroll
  for (int t = 0; t < 7; ++t) {
    int job = wave + t * 8;
    int mi = job / (LP / 16);
    int nj = job % (LP / 16);
    v8f acc = {};
#pragma unroll
    for (int kk = 0; kk < LP; kk += 32)
      acc = wmma_bf(a_frag_lds_bf16(comb, LP, mi * 16, kk),
                    bT_frag_bf16(Wg1t, LP, nj * 16, kk), acc);
    int n = nj * 16 + (lane & 15);
    float w2 = wg2p[n];
    float b1 = bg1p[n];
#pragma unroll
    for (int r = 0; r < 8; ++r) {
      int row = mi * 16 + r + 8 * hw;          // = s*16 + q-row
      atomicAdd(&energy[row], fmaxf(acc[r] + b1, 0.0f) * w2);  // ds_add_f32
    }
  }
  __syncthreads();

  // ---- Stage 3: softmax over the 4 streams -> gate weights
  if (tid < 16) {
    int m = tid;
    float bb2 = bg2v[0];
    float e0 = energy[m] + bb2, e1 = energy[16 + m] + bb2;
    float e2 = energy[32 + m] + bb2, e3 = energy[48 + m] + bb2;
    float mx = fmaxf(fmaxf(e0, e1), fmaxf(e2, e3));
    float x0 = __expf(e0 - mx), x1 = __expf(e1 - mx);
    float x2 = __expf(e2 - mx), x3 = __expf(e3 - mx);
    float inv = 1.0f / (x0 + x1 + x2 + x3);
    gw[m] = x0 * inv; gw[16 + m] = x1 * inv;
    gw[32 + m] = x2 * inv; gw[48 + m] = x3 * inv;
  }
  __syncthreads();

  // ---- Stage 4: gated scores + scale + mask
  const float rs = 0.17677669529663688f;       // 1/sqrt(d)
  for (int idx = tid; idx < 16 * LP; idx += 256) {
    int m = idx / LP, col = idx % LP;
    int qrow = q0 + m;
    float v;
    if (col < Ll && qrow < Ll) {
      float a = 0.0f;
#pragma unroll
      for (int s = 0; s < 4; ++s)
        a += gw[s * 16 + m] * (float)comb[(s * 16 + m) * LP + col];
      v = a * rs + mask[((long)b * Ll + qrow) * Ll + col];
    } else {
      v = -INFINITY;
    }
    scores[idx] = v;
  }
  __syncthreads();

  // ---- Stage 5: softmax over keys (16 threads per q row)
  {
    int m = tid >> 4, j = tid & 15;
    float mx = -INFINITY;
    for (int c = j; c < LP; c += 16) mx = fmaxf(mx, scores[m * LP + c]);
    red[tid] = mx;
    __syncthreads();
    if (j == 0) {
      float r = red[m * 16];
#pragma unroll
      for (int t = 1; t < 16; ++t) r = fmaxf(r, red[m * 16 + t]);
      rowstat[m] = r;
    }
    __syncthreads();
    float rm = rowstat[m];
    float sum = 0.0f;
    for (int c = j; c < LP; c += 16) {
      float sc = scores[m * LP + c];
      float p  = (sc > -1e37f) ? __expf(sc - rm) : 0.0f;
      scores[m * LP + c] = p;
      sum += p;
    }
    __syncthreads();
    red[tid] = sum;
    __syncthreads();
    if (j == 0) {
      float r = 0.0f;
#pragma unroll
      for (int t = 0; t < 16; ++t) r += red[m * 16 + t];
      rowstat[m] = r;
    }
    __syncthreads();
    float denom = rowstat[m];
    float inv = (denom > 0.0f) ? 1.0f / denom : 0.0f;
    for (int c = j; c < LP; c += 16)
      probsb[m * LP + c] = (__bf16)(scores[m * LP + c] * inv);
  }
  __syncthreads();

  // ---- Stage 6: ctx = probs(16xLP) @ v(LPx32); 2 col tiles, 7 K-steps
  if (wave < 2) {
    int n0 = wave * 16;
    v8f acc = {};
#pragma unroll
    for (int kk = 0; kk < LP; kk += 32)
      acc = wmma_bf(a_frag_lds_bf16(probsb, LP, 0, kk),
                    b_frag_bf16(V + rowBase * Dd + hcol, Dd, kk, n0), acc);
#pragma unroll
    for (int r = 0; r < 8; ++r) {
      int qr = q0 + r + 8 * hw;
      if (qr < Ll)
        CTXb[(rowBase + qr) * Dd + hcol + n0 + (lane & 15)] = (__bf16)acc[r];
    }
  }
}

// ===========================================================================
// Kernel 3: out = LayerNorm(ctx @ Wd + bd + input) * g + b   (LN fused)
// ===========================================================================
__global__ __launch_bounds__(256) void out_ln_kernel(
    const __bf16* __restrict__ CTXb, const __bf16* __restrict__ WdT,
    const float* __restrict__ bd,  const float* __restrict__ X,
    const float* __restrict__ g,   const float* __restrict__ bb,
    float* __restrict__ out) {
  __shared__ float hrow[16 * Dd];
  __shared__ float red[256];
  __shared__ float stat[16];

  int tid  = threadIdx.x;
  int wave = tid >> 5;
  int lane = tid & 31;
  int hw   = lane >> 4;
  long r0  = (long)blockIdx.x * 16;

#pragma unroll
  for (int t = 0; t < 2; ++t) {
    int n0 = (wave * 2 + t) * 16;
    v8f acc = {};
#pragma unroll
    for (int kk = 0; kk < Dd; kk += 32)
      acc = wmma_bf(a_frag_bf16(CTXb, Dd, r0, kk),
                    bT_frag_bf16(WdT, Dd, n0, kk), acc);
    int col = n0 + (lane & 15);
#pragma unroll
    for (int r = 0; r < 8; ++r) {
      int row = r + 8 * hw;
      hrow[row * Dd + col] = acc[r] + bd[col] + X[(r0 + row) * Dd + col];
    }
  }
  __syncthreads();

  int m = tid >> 4, j = tid & 15;
  float s = 0.0f;
  for (int c = j; c < Dd; c += 16) s += hrow[m * Dd + c];
  red[tid] = s;
  __syncthreads();
  if (j == 0) {
    float r = 0.0f;
#pragma unroll
    for (int t = 0; t < 16; ++t) r += red[m * 16 + t];
    stat[m] = r * (1.0f / Dd);
  }
  __syncthreads();
  float mu = stat[m];
  float vs = 0.0f;
  for (int c = j; c < Dd; c += 16) {
    float d = hrow[m * Dd + c] - mu;
    vs += d * d;
  }
  __syncthreads();
  red[tid] = vs;
  __syncthreads();
  if (j == 0) {
    float r = 0.0f;
#pragma unroll
    for (int t = 0; t < 16; ++t) r += red[m * 16 + t];
    stat[m] = r * (1.0f / Dd);
  }
  __syncthreads();
  float inv = rsqrtf(stat[m] + 1e-12f);
  for (int c = j; c < Dd; c += 16) {
    float hv = hrow[m * Dd + c];
    out[(r0 + m) * Dd + c] = (hv - mu) * inv * g[c] + bb[c];
  }
}

// ===========================================================================
// Host launcher
// ===========================================================================
extern "C" void kernel_launch(void* const* d_in, const int* in_sizes, int n_in,
                              void* d_out, int out_size, void* d_ws, size_t ws_size,
                              hipStream_t stream) {
  (void)in_sizes; (void)n_in; (void)out_size; (void)ws_size;

  const float* input = (const float*)d_in[0];
  const float* attr  = (const float*)d_in[1];
  const float* pos   = (const float*)d_in[2];
  const float* mask  = (const float*)d_in[3];
  const float* Wq  = (const float*)d_in[4];  const float* bq  = (const float*)d_in[5];
  const float* Wk  = (const float*)d_in[6];  const float* bk  = (const float*)d_in[7];
  const float* Wv  = (const float*)d_in[8];  const float* bv  = (const float*)d_in[9];
  const float* Wqp = (const float*)d_in[10]; const float* bqp = (const float*)d_in[11];
  const float* Wkp = (const float*)d_in[12]; const float* bkp = (const float*)d_in[13];
  const float* Wqa = (const float*)d_in[14]; const float* bqa = (const float*)d_in[15];
  const float* Wka = (const float*)d_in[16]; const float* bka = (const float*)d_in[17];
  const float* Wg1 = (const float*)d_in[18]; const float* bg1 = (const float*)d_in[19];
  const float* Wg2 = (const float*)d_in[20]; const float* bg2 = (const float*)d_in[21];
  const float* Wd  = (const float*)d_in[22]; const float* bd  = (const float*)d_in[23];
  const float* lng = (const float*)d_in[24]; const float* lnb = (const float*)d_in[25];

  const long MT = (long)MROWS * Dd;          // elements per activation buffer
  __bf16* wsb  = (__bf16*)d_ws;
  __bf16* Qb   = wsb + 0 * MT;
  __bf16* Kb   = wsb + 1 * MT;
  __bf16* Vb   = wsb + 2 * MT;
  __bf16* QPb  = wsb + 3 * MT;
  __bf16* KPb  = wsb + 4 * MT;
  __bf16* QAb  = wsb + 5 * MT;               // 2 streams, stride MT
  __bf16* KAb  = wsb + 7 * MT;               // 2 streams, stride MT
  __bf16* CTXb = wsb + 9 * MT;
  __bf16* Wg1t = wsb + 10 * MT;              // LP*LP
  __bf16* WdT  = Wg1t + (long)LP * LP;       // Dd*Dd
  float*  fext = (float*)(WdT + (long)Dd * Dd);
  float*  bg1p = fext;                       // LP
  float*  wg2p = fext + LP;                  // LP

  dim3 blk(256);

  prep_gate_kernel<<<(LP * LP + 255) / 256, blk, 0, stream>>>(Wg1, bg1, Wg2,
                                                              Wg1t, bg1p, wg2p);
  prep_wd_kernel<<<(Dd * Dd) / 256, blk, 0, stream>>>(Wd, WdT);

  dim3 gproj(Dd / 16, MROWS / 128);          // (16, 100)
  gemm_bias_kernel<<<gproj, blk, 0, stream>>>(input, Wq,  bq,  Qb);
  gemm_bias_kernel<<<gproj, blk, 0, stream>>>(input, Wk,  bk,  Kb);
  gemm_bias_kernel<<<gproj, blk, 0, stream>>>(input, Wv,  bv,  Vb);
  gemm_bias_kernel<<<gproj, blk, 0, stream>>>(pos,   Wqp, bqp, QPb);
  gemm_bias_kernel<<<gproj, blk, 0, stream>>>(pos,   Wkp, bkp, KPb);
  for (int f = 0; f < Ff; ++f) {
    gemm_bias_kernel<<<gproj, blk, 0, stream>>>(attr + (long)f * MT,
                                                Wqa + (long)f * Dd * Dd,
                                                bqa + (long)f * Dd, QAb + (long)f * MT);
    gemm_bias_kernel<<<gproj, blk, 0, stream>>>(attr + (long)f * MT,
                                                Wka + (long)f * Dd * Dd,
                                                bka + (long)f * Dd, KAb + (long)f * MT);
  }

  dim3 gattn((Ll + 15) / 16, Hh, Bb);        // (13, 8, 64)
  attn_fused_kernel<<<gattn, blk, 0, stream>>>(Qb, Kb, Vb, QPb, KPb, QAb, KAb, mask,
                                               Wg1t, bg1p, wg2p, bg2, CTXb);

  out_ln_kernel<<<MROWS / 16, blk, 0, stream>>>(CTXb, WdT, bd, input, lng, lnb,
                                                (float*)d_out);
}
